// GraphormerLayer_21492016349938
// MI455X (gfx1250) — matrix operand order repaired and checked by reading, not compiled, not run on hardware
//
#include <hip/hip_runtime.h>
#include <hip/hip_bf16.h>
#include <math.h>

#define N_TOK   4096
#define DMODEL  512
#define NHEADS  8
#define HDIM    64
#define FFNDIM  2048
#define LN_EPS  1e-5f
#define PLS_LD  72   // padded LDS row stride (bf16): 144B = 36 banks; 36*l mod 64 distinct

typedef __bf16 bf16;
typedef __attribute__((ext_vector_type(16))) __bf16 v16bf;
typedef __attribute__((ext_vector_type(8)))  __bf16 v8bf;
typedef __attribute__((ext_vector_type(8)))  float  v8f;

// ---------------------------------------------------------------- WMMA core
__device__ __forceinline__ v8f wmma_bf16(v16bf a, v16bf b, v8f c) {
  return __builtin_amdgcn_wmma_f32_16x16x32_bf16(
      false, a, false, b, (short)0, c, false, false);
}

__device__ __forceinline__ v16bf pack16(v8bf lo, v8bf hi) {
  v16bf r;
#pragma unroll
  for (int i = 0; i < 8; ++i) { r[i] = lo[i]; r[i + 8] = hi[i]; }
  return r;
}

// A fragment: rows row0..row0+15 x K k0..k0+31, row-major source (ld elems).
__device__ __forceinline__ v16bf load_a_frag(const bf16* base, int ld,
                                             int row0, int k0) {
  const int lane = threadIdx.x & 31;
  const int half = lane >> 4, l = lane & 15;
  const bf16* p = base + (size_t)(row0 + l) * ld + k0 + half * 8;
  return pack16(*(const v8bf*)p, *(const v8bf*)(p + 16));
}

// B fragment: K k0..k0+31 x cols n0..n0+15; source stored (col, k) row-major.
__device__ __forceinline__ v16bf load_b_frag(const bf16* base, int ld,
                                             int n0, int k0) {
  const int lane = threadIdx.x & 31;
  const int half = lane >> 4, l = lane & 15;
  const bf16* p = base + (size_t)(n0 + l) * ld + k0 + half * 8;
  return pack16(*(const v8bf*)p, *(const v8bf*)(p + 16));
}

// ---------------------------------------------------------------- reductions
__device__ __forceinline__ float wave_sum32(float v) {
  v += __shfl_xor(v, 1, 32);  v += __shfl_xor(v, 2, 32);
  v += __shfl_xor(v, 4, 32);  v += __shfl_xor(v, 8, 32);
  v += __shfl_xor(v, 16, 32); return v;
}
__device__ __forceinline__ float half16_max(float v) {
  v = fmaxf(v, __shfl_xor(v, 1, 32)); v = fmaxf(v, __shfl_xor(v, 2, 32));
  v = fmaxf(v, __shfl_xor(v, 4, 32)); v = fmaxf(v, __shfl_xor(v, 8, 32));
  return v;
}
__device__ __forceinline__ float half16_sum(float v) {
  v += __shfl_xor(v, 1, 32); v += __shfl_xor(v, 2, 32);
  v += __shfl_xor(v, 4, 32); v += __shfl_xor(v, 8, 32);
  return v;
}

__device__ __forceinline__ float gelu_exact(float v) {
  return 0.5f * v * (1.0f + erff(v * 0.70710678118654752f));
}

// ---------------------------------------------------------------- weight prep
__global__ void wprep_kernel(const float* __restrict__ W, bf16* __restrict__ WT,
                             int K, int Nout) {
  size_t i = (size_t)blockIdx.x * 256 + threadIdx.x;
  if (i >= (size_t)K * Nout) return;
  int n = (int)(i / K), kk = (int)(i % K);
  WT[i] = (bf16)W[(size_t)kk * Nout + n];
}

// V (N, DMODEL) -> VT (DMODEL rows, N cols)
__global__ void vt_kernel(const bf16* __restrict__ v, bf16* __restrict__ vt) {
  int n = blockIdx.x * 256 + threadIdx.x;
  int c = blockIdx.y;
  vt[(size_t)c * N_TOK + n] = v[(size_t)n * DMODEL + c];
}

// ---------------------------------------------------------------- layernorm
__global__ void ln_kernel(const float* __restrict__ x, const float* __restrict__ g,
                          const float* __restrict__ b, bf16* __restrict__ h) {
  const int w = threadIdx.x >> 5, lane = threadIdx.x & 31;
  const int row = blockIdx.x * 8 + w;
  const float* xr = x + (size_t)row * DMODEL;
  float xv[16], s = 0.f;
#pragma unroll
  for (int i = 0; i < 16; ++i) { xv[i] = xr[lane + i * 32]; s += xv[i]; }
  const float mu = wave_sum32(s) * (1.0f / DMODEL);
  float vs = 0.f;
#pragma unroll
  for (int i = 0; i < 16; ++i) { float d = xv[i] - mu; vs += d * d; }
  const float rstd = rsqrtf(wave_sum32(vs) * (1.0f / DMODEL) + LN_EPS);
  bf16* hr = h + (size_t)row * DMODEL;
#pragma unroll
  for (int i = 0; i < 16; ++i) {
    int c = lane + i * 32;
    hr[c] = (bf16)((xv[i] - mu) * rstd * g[c] + b[c]);
  }
}

// ---------------------------------------------------------------- WMMA GEMM
// EPI: 0 -> store bf16 (scaled); 1 -> gelu, bf16; 2 -> +resid, f32.
template <int EPI>
__global__ void gemm_kernel(const bf16* __restrict__ A, const bf16* __restrict__ WT,
                            const float* __restrict__ bias,
                            const float* __restrict__ resid,
                            bf16* __restrict__ outB, float* __restrict__ outF,
                            int M, int Nout, int K, float oscale) {
  const int w  = threadIdx.x >> 5;
  const int m0 = blockIdx.y * 128 + (w >> 1) * 32;
  const int n0 = blockIdx.x * 64 + (w & 1) * 32;
  const int lane = threadIdx.x & 31, half = lane >> 4, l = lane & 15;

  v8f c[2][2] = {};
  for (int k0 = 0; k0 < K; k0 += 32) {
    if (k0 + 64 <= K)
      __builtin_prefetch(A + (size_t)(m0 + lane) * K + k0 + 32, 0, 1);
    v16bf a0 = load_a_frag(A, K, m0,      k0);
    v16bf a1 = load_a_frag(A, K, m0 + 16, k0);
    v16bf b0 = load_b_frag(WT, K, n0,      k0);
    v16bf b1 = load_b_frag(WT, K, n0 + 16, k0);
    c[0][0] = wmma_bf16(a0, b0, c[0][0]);
    c[0][1] = wmma_bf16(a0, b1, c[0][1]);
    c[1][0] = wmma_bf16(a1, b0, c[1][0]);
    c[1][1] = wmma_bf16(a1, b1, c[1][1]);
  }
#pragma unroll
  for (int i = 0; i < 2; ++i)
#pragma unroll
    for (int j = 0; j < 2; ++j)
#pragma unroll
      for (int r = 0; r < 8; ++r) {
        const int row = m0 + i * 16 + r + half * 8;
        const int col = n0 + j * 16 + l;
        float v = (c[i][j][r] + bias[col]) * oscale;
        if (EPI == 1) v = gelu_exact(v);
        if (EPI == 2) outF[(size_t)row * Nout + col] = v + resid[(size_t)row * Nout + col];
        else          outB[(size_t)row * Nout + col] = (bf16)v;
      }
}

// ---------------------------------------------------------------- flash attention
// Block = 8 waves = the 8 heads of ONE 16-query tile (bias lines fully shared).
// Key block = 64: softmax reductions / rescales / LDS round-trips amortized 2x.
// Q is pre-scaled by 1/sqrt(d) in its projection.
__global__ void __launch_bounds__(256, 1)
attn_kernel(const bf16* __restrict__ q, const bf16* __restrict__ k,
            const bf16* __restrict__ vt,
            const float* __restrict__ bias, bf16* __restrict__ o) {
  __shared__ __align__(16) bf16 pls[8][16 * PLS_LD];
  const int w = threadIdx.x >> 5;            // wave id == head id
  const int lane = threadIdx.x & 31, half = lane >> 4, l = lane & 15;
  const int h  = w;
  const int n0 = blockIdx.x << 4;

  const bf16* qh  = q  + h * HDIM;
  const bf16* kh  = k  + h * HDIM;
  const bf16* vth = vt + (size_t)h * HDIM * N_TOK;
  bf16* plw = &pls[w][0];

  // single lane-dependent bias base; (r, tile) offsets are compile-time imms
  const float* bptr = bias + ((size_t)(n0 + half * 8) * N_TOK + l) * NHEADS + h;

  const v16bf aq0 = load_a_frag(qh, DMODEL, n0, 0);
  const v16bf aq1 = load_a_frag(qh, DMODEL, n0, 32);

  v8f acc[4] = {};
  float mrow[8], lrow[8];
#pragma unroll
  for (int r = 0; r < 8; ++r) { mrow[r] = -INFINITY; lrow[r] = 0.f; }

  for (int m0 = 0; m0 < N_TOK; m0 += 64) {
    // 32 bias loads, all immediate offsets off one pointer -> one clause
    float b[4][8];
#pragma unroll
    for (int t = 0; t < 4; ++t)
#pragma unroll
      for (int r = 0; r < 8; ++r)
        b[t][r] = bptr[r * (N_TOK * NHEADS) + t * (16 * NHEADS)];
    bptr += 64 * NHEADS;

    // S = Q K^T for 4 16-col tiles (8 WMMAs)
    v8f s[4];
#pragma unroll
    for (int t = 0; t < 4; ++t) {
      v8f z = {};
      z = wmma_bf16(aq0, load_b_frag(kh, DMODEL, m0 + t * 16, 0),  z);
      s[t] = wmma_bf16(aq1, load_b_frag(kh, DMODEL, m0 + t * 16, 32), z);
    }

    float p[4][8];
#pragma unroll
    for (int t = 0; t < 4; ++t)
#pragma unroll
      for (int r = 0; r < 8; ++r)
        p[t][r] = s[t][r] + b[t][r];

#pragma unroll
    for (int r = 0; r < 8; ++r) {
      float rm = fmaxf(fmaxf(p[0][r], p[1][r]), fmaxf(p[2][r], p[3][r]));
      rm = half16_max(rm);
      const float mnew  = fmaxf(mrow[r], rm);
      const float alpha = __expf(mrow[r] - mnew);
#pragma unroll
      for (int t = 0; t < 4; ++t) p[t][r] = __expf(p[t][r] - mnew);
      const float rs = half16_sum((p[0][r] + p[1][r]) + (p[2][r] + p[3][r]));
      lrow[r] = lrow[r] * alpha + rs;
      mrow[r] = mnew;
#pragma unroll
      for (int j = 0; j < 4; ++j) acc[j][r] *= alpha;
#pragma unroll
      for (int t = 0; t < 4; ++t)
        plw[(r + half * 8) * PLS_LD + t * 16 + l] = (bf16)p[t][r];
    }
    // per-wave LDS slice; DS ops are in-order within a wave -> no block barrier
    asm volatile("s_wait_dscnt 0x0" ::: "memory");
    const v16bf pa0 = load_a_frag((const bf16*)plw, PLS_LD, 0, 0);
    const v16bf pa1 = load_a_frag((const bf16*)plw, PLS_LD, 0, 32);
#pragma unroll
    for (int j = 0; j < 4; ++j) {
      acc[j] = wmma_bf16(pa0, load_b_frag(vth, N_TOK, j * 16, m0),      acc[j]);
      acc[j] = wmma_bf16(pa1, load_b_frag(vth, N_TOK, j * 16, m0 + 32), acc[j]);
    }
  }
#pragma unroll
  for (int j = 0; j < 4; ++j)
#pragma unroll
    for (int r = 0; r < 8; ++r) {
      const int n = n0 + r + half * 8;
      const int col = h * HDIM + j * 16 + l;
      o[(size_t)n * DMODEL + col] = (bf16)(acc[j][r] / lrow[r]);
    }
}

// ---------------------------------------------------------------- launch
extern "C" void kernel_launch(void* const* d_in, const int* in_sizes, int n_in,
                              void* d_out, int out_size, void* d_ws, size_t ws_size,
                              hipStream_t stream) {
  (void)in_sizes; (void)n_in; (void)out_size; (void)ws_size;
  const float* x     = (const float*)d_in[0];
  const float* abias = (const float*)d_in[1];
  const float* wq = (const float*)d_in[2];  const float* bq  = (const float*)d_in[3];
  const float* wk = (const float*)d_in[4];  const float* bk  = (const float*)d_in[5];
  const float* wv = (const float*)d_in[6];  const float* bv  = (const float*)d_in[7];
  const float* wo = (const float*)d_in[8];  const float* bo  = (const float*)d_in[9];
  const float* w1 = (const float*)d_in[10]; const float* b1  = (const float*)d_in[11];
  const float* w2 = (const float*)d_in[12]; const float* b2  = (const float*)d_in[13];
  const float* g1 = (const float*)d_in[14]; const float* be1 = (const float*)d_in[15];
  const float* g2 = (const float*)d_in[16]; const float* be2 = (const float*)d_in[17];
  float* out = (float*)d_out;

  unsigned char* ws = (unsigned char*)d_ws;
  size_t off = 0;
  auto take = [&](size_t bytes) -> void* {
    void* p = ws + off; off += (bytes + 255) & ~(size_t)255; return p;
  };
  bf16* h_bf = (bf16*)take((size_t)N_TOK * DMODEL * 2);
  bf16* wqT  = (bf16*)take((size_t)DMODEL * DMODEL * 2);
  bf16* wkT  = (bf16*)take((size_t)DMODEL * DMODEL * 2);
  bf16* wvT  = (bf16*)take((size_t)DMODEL * DMODEL * 2);
  bf16* woT  = (bf16*)take((size_t)DMODEL * DMODEL * 2);
  bf16* w1T  = (bf16*)take((size_t)FFNDIM * DMODEL * 2);
  bf16* w2T  = (bf16*)take((size_t)DMODEL * FFNDIM * 2);
  bf16* qb   = (bf16*)take((size_t)N_TOK * DMODEL * 2);
  bf16* kb   = (bf16*)take((size_t)N_TOK * DMODEL * 2);
  bf16* vb   = (bf16*)take((size_t)N_TOK * DMODEL * 2);
  bf16* vtb  = (bf16*)take((size_t)N_TOK * DMODEL * 2);
  bf16* ob   = (bf16*)take((size_t)N_TOK * DMODEL * 2);
  float* x2  = (float*)take((size_t)N_TOK * DMODEL * 4);
  bf16* h2   = (bf16*)take((size_t)N_TOK * DMODEL * 2);
  bf16* mid  = (bf16*)take((size_t)N_TOK * FFNDIM * 2);

  auto wlaunch = [&](const float* W, bf16* WT, int K, int Nout) {
    size_t total = (size_t)K * Nout;
    wprep_kernel<<<dim3((unsigned)((total + 255) / 256)), dim3(256), 0, stream>>>(W, WT, K, Nout);
  };
  wlaunch(wq, wqT, DMODEL, DMODEL);
  wlaunch(wk, wkT, DMODEL, DMODEL);
  wlaunch(wv, wvT, DMODEL, DMODEL);
  wlaunch(wo, woT, DMODEL, DMODEL);
  wlaunch(w1, w1T, DMODEL, FFNDIM);
  wlaunch(w2, w2T, FFNDIM, DMODEL);

  ln_kernel<<<dim3(N_TOK / 8), dim3(256), 0, stream>>>(x, g1, be1, h_bf);

  const dim3 gqkv(DMODEL / 64, N_TOK / 128);
  // Q pre-scaled by 1/sqrt(HDIM)
  gemm_kernel<0><<<gqkv, dim3(256), 0, stream>>>(h_bf, wqT, bq, nullptr, qb, nullptr, N_TOK, DMODEL, DMODEL, 0.125f);
  gemm_kernel<0><<<gqkv, dim3(256), 0, stream>>>(h_bf, wkT, bk, nullptr, kb, nullptr, N_TOK, DMODEL, DMODEL, 1.0f);
  gemm_kernel<0><<<gqkv, dim3(256), 0, stream>>>(h_bf, wvT, bv, nullptr, vb, nullptr, N_TOK, DMODEL, DMODEL, 1.0f);

  vt_kernel<<<dim3(N_TOK / 256, DMODEL), dim3(256), 0, stream>>>(vb, vtb);

  // one block per query tile; 8 waves = 8 heads sharing bias cache lines
  attn_kernel<<<dim3(N_TOK / 16), dim3(256), 0, stream>>>(qb, kb, vtb, abias, ob);

  gemm_kernel<2><<<gqkv, dim3(256), 0, stream>>>(ob, woT, bo, x, nullptr, x2, N_TOK, DMODEL, DMODEL, 1.0f);

  ln_kernel<<<dim3(N_TOK / 8), dim3(256), 0, stream>>>(x2, g2, be2, h2);

  gemm_kernel<1><<<dim3(FFNDIM / 64, N_TOK / 128), dim3(256), 0, stream>>>(h2, w1T, b1, nullptr, mid, nullptr, N_TOK, FFNDIM, DMODEL, 1.0f);

  gemm_kernel<2><<<dim3(DMODEL / 64, N_TOK / 128), dim3(256), 0, stream>>>(mid, w2T, b2, x2, nullptr, out, N_TOK, DMODEL, FFNDIM, 1.0f);
}
